// GraphSAGEEdgeRegressor_20289425506996
// MI455X (gfx1250) — matrix operand order repaired and checked by reading, not compiled, not run on hardware
//
#include <hip/hip_runtime.h>
#include <hip/hip_bf16.h>
#include <stdint.h>

typedef __attribute__((ext_vector_type(16))) __bf16 v16bf;
typedef __attribute__((ext_vector_type(8)))  float  v8f;

#define HID 128
#define BN_EPS 1e-5f

// ---------------------------------------------------------------------------
// WMMA wrapper: D = A(16x32 bf16) * B(32x16 bf16) + C(16x16 f32)
// ---------------------------------------------------------------------------
__device__ inline v8f wmma_bf16(v16bf a, v16bf b, v8f c) {
  return __builtin_amdgcn_wmma_f32_16x16x32_bf16(
      /*neg_a=*/false, a, /*neg_b=*/false, b,
      /*c_mod=*/(short)0, c, /*reuse_a=*/false, /*reuse_b=*/false);
}

// ---------------------------------------------------------------------------
// Async global->LDS staging (CDNA5 GLOBAL_LOAD_ASYNC_TO_LDS_B128, ASYNCcnt).
// Each lane copies 16-byte chunks directly into LDS, bypassing VGPRs.
// bytes must be a multiple of 16; both pointers 16B aligned.
// ---------------------------------------------------------------------------
__device__ inline void async_stage_to_lds(const void* gsrc, void* ldsDst, int bytes) {
  uint32_t ldsBase = (uint32_t)(uintptr_t)ldsDst;   // low 32 bits = LDS offset
  const char* g = (const char*)gsrc;
  for (int off = threadIdx.x * 16; off < bytes; off += blockDim.x * 16) {
    uint32_t laddr = ldsBase + off;
    uint64_t gaddr = (uint64_t)(uintptr_t)(g + off);
    asm volatile("global_load_async_to_lds_b128 %0, %1, off"
                 :: "v"(laddr), "v"(gaddr) : "memory");
  }
  asm volatile("s_wait_asynccnt 0x0" ::: "memory");
}

// ---------------------------------------------------------------------------
// A-matrix loaders (ISA 16-bit A 16x32 layout):
//   lane l owns row m=l&15; half = l>>4
//   v16bf positions 0..7  -> K = ktile*32 + half*8 + {0..7}
//   v16bf positions 8..15 -> K = ktile*32 + half*8 + 16 + {0..7}
// ---------------------------------------------------------------------------
__device__ inline v16bf loadA_f32(const float* row, int ktile, float scale) {
  int kh = ((threadIdx.x >> 4) & 1) * 8;
  const float* p = row + ktile * 32 + kh;
  v16bf a;
#pragma unroll
  for (int i = 0; i < 8; ++i) {
    a[i]     = (__bf16)(p[i] * scale);
    a[i + 8] = (__bf16)(p[i + 16] * scale);
  }
  return a;
}

__device__ inline v16bf loadA_bf16(const __bf16* row, int ktile) {
  int kh = ((threadIdx.x >> 4) & 1) * 8;
  const __bf16* p = row + ktile * 32 + kh;
  v16bf a;
#pragma unroll
  for (int i = 0; i < 8; ++i) {
    a[i]     = p[i];
    a[i + 8] = p[i + 16];
  }
  return a;
}

// bf16 row with per-channel affine (folded BatchNorm) applied on load
__device__ inline v16bf loadA_bf16_bn(const __bf16* row, int ktile,
                                      const float* sc, const float* sh) {
  int kh = ((threadIdx.x >> 4) & 1) * 8;
  int base = ktile * 32 + kh;
  v16bf a;
#pragma unroll
  for (int i = 0; i < 8; ++i) {
    int k0 = base + i, k1 = base + 16 + i;
    a[i]     = (__bf16)((float)row[k0] * sc[k0] + sh[k0]);
    a[i + 8] = (__bf16)((float)row[k1] * sc[k1] + sh[k1]);
  }
  return a;
}

// ---------------------------------------------------------------------------
// Pre-swizzle a [K x (nct*16)] f32 weight matrix (global or LDS source) into
// the exact per-lane WMMA B layout: lane l owns col n=l&15,
// K = (p>>3)*16 + (l>>4)*8 + (p&7) within each 32-K tile -> one contiguous v16bf per
// (ktile,ctile,lane). K padded to Kpad (mult of 32) with zeros.
// ---------------------------------------------------------------------------
__device__ inline void fill_B_lds(const float* __restrict__ W, __bf16* lds,
                                  int kpad, int kreal, int nct) {
  int total = (kpad / 32) * nct * 32 * 16;
  int ldw = nct * 16;
  for (int idx = threadIdx.x; idx < total; idx += blockDim.x) {
    int p    = idx & 15;
    int lane = (idx >> 4) & 31;
    int ctkt = idx >> 9;
    int ct   = ctkt % nct;
    int kt   = ctkt / nct;
    int k    = kt * 32 + (p >> 3) * 16 + (lane >> 4) * 8 + (p & 7);
    int n    = ct * 16 + (lane & 15);
    float v  = (k < kreal) ? W[(size_t)k * ldw + n] : 0.0f;
    lds[idx] = (__bf16)v;
  }
}

__device__ inline v16bf loadB(const __bf16* lds, int kt, int ct, int nct) {
  const __bf16* p = lds + ((size_t)(kt * nct + ct) * 32 + (threadIdx.x & 31)) * 16;
  v16bf b;
#pragma unroll
  for (int i = 0; i < 16; ++i) b[i] = p[i];
  return b;
}

// ---------------------------------------------------------------------------
// Utility kernels
// ---------------------------------------------------------------------------
__global__ void zero_kernel(float* p, long n) {
  long i = (long)blockIdx.x * blockDim.x + threadIdx.x;
  long s = (long)gridDim.x * blockDim.x;
  for (; i < n; i += s) p[i] = 0.0f;
}

__global__ void count_kernel(const int* __restrict__ dst, float* cnt, int E) {
  int i = blockIdx.x * blockDim.x + threadIdx.x;
  if (i < E) atomicAdd(&cnt[dst[i]], 1.0f);
}

template <int F>
__global__ void scatter_add_kernel(const float* __restrict__ feat,
                                   const int* __restrict__ src,
                                   const int* __restrict__ dst,
                                   float* acc, int E) {
  constexpr int SH = (F == 32) ? 5 : 7;
  long total = (long)E * F;
  long i = (long)blockIdx.x * blockDim.x + threadIdx.x;
  long s = (long)gridDim.x * blockDim.x;
  for (; i < total; i += s) {
    int e = (int)(i >> SH);
    int c = (int)(i & (F - 1));
    atomicAdd(&acc[(size_t)dst[e] * F + c], feat[(size_t)src[e] * F + c]);
  }
}

// mean/var -> (scale, shift) so that bn(x) = x*scale + shift
__global__ void bn_finalize_kernel(const float* sum, const float* sq,
                                   const float* g, const float* b,
                                   float* scale, float* shift, float count) {
  int c = threadIdx.x;
  if (c < HID) {
    float m  = sum[c] / count;
    float v  = sq[c] / count - m * m;
    float rs = rsqrtf(v + BN_EPS);
    float sc = g[c] * rs;
    scale[c] = sc;
    shift[c] = b[c] - m * sc;
  }
}

template <typename OutT>
__global__ void bn_apply_relu_kernel(const float* __restrict__ h,
                                     const float* __restrict__ scale,
                                     const float* __restrict__ shift,
                                     OutT* out, long n) {
  long i = (long)blockIdx.x * blockDim.x + threadIdx.x;
  long s = (long)gridDim.x * blockDim.x;
  for (; i < n; i += s) {
    int c = (int)(i & (HID - 1));
    float v = fmaxf(h[i] * scale[c] + shift[c], 0.0f);
    out[i] = (OutT)v;
  }
}

// ---------------------------------------------------------------------------
// SAGE layer GEMM: hpre[16 x 128] = mean(16xF)@Wl + x(16xF)@Wr + b
// one wave per 16-node tile; 8 waves / block share LDS-resident weights.
// F==128 stages the f32 weights via async global->LDS DMA first.
// ---------------------------------------------------------------------------
template <int F>
__global__ __launch_bounds__(256) void sage_gemm_kernel(
    const float* __restrict__ agg, const float* __restrict__ cnt,
    const float* __restrict__ xin,
    const float* __restrict__ Wl, const float* __restrict__ bias,
    const float* __restrict__ Wr,
    float* __restrict__ hpre, float* bn_sum, float* bn_sq, int nTiles) {
  extern __shared__ char smemraw[];
  constexpr int KT = F / 32;
  __bf16* ldsBl = (__bf16*)smemraw;
  __bf16* ldsBr = ldsBl + (size_t)F * HID;
  if constexpr (F == 128) {
    float* stage = (float*)(ldsBr + (size_t)F * HID);
    async_stage_to_lds(Wl, stage, F * HID * 4);
    async_stage_to_lds(Wr, stage + (size_t)F * HID, F * HID * 4);
    __syncthreads();
    fill_B_lds(stage, ldsBl, F, F, 8);
    fill_B_lds(stage + (size_t)F * HID, ldsBr, F, F, 8);
  } else {
    fill_B_lds(Wl, ldsBl, F, F, 8);
    fill_B_lds(Wr, ldsBr, F, F, 8);
  }
  __syncthreads();

  int wave = threadIdx.x >> 5;
  int lane = threadIdx.x & 31;
  int tile = blockIdx.x * 8 + wave;
  if (tile >= nTiles) return;

  int m = lane & 15;
  int half = lane >> 4;
  int node = tile * 16 + m;
  float ic = 1.0f / fmaxf(cnt[node], 1.0f);
  const float* arow = agg + (size_t)node * F;
  const float* xrow = xin + (size_t)node * F;

  v16bf A1[KT], A2[KT];
#pragma unroll
  for (int kt = 0; kt < KT; ++kt) {
    A1[kt] = loadA_f32(arow, kt, ic);
    A2[kt] = loadA_f32(xrow, kt, 1.0f);
  }

  for (int ct = 0; ct < 8; ++ct) {
    float bv = bias[ct * 16 + m];
    v8f c;
#pragma unroll
    for (int r = 0; r < 8; ++r) c[r] = bv;
#pragma unroll
    for (int kt = 0; kt < KT; ++kt) {
      c = wmma_bf16(A1[kt], loadB(ldsBl, kt, ct, 8), c);
      c = wmma_bf16(A2[kt], loadB(ldsBr, kt, ct, 8), c);
    }
    float s = 0.f, q = 0.f;
#pragma unroll
    for (int r = 0; r < 8; ++r) {
      float v = c[r];
      s += v; q += v * v;
      hpre[(size_t)(tile * 16 + r + 8 * half) * HID + ct * 16 + m] = v;
    }
    s += __shfl_xor(s, 16, 32);
    q += __shfl_xor(q, 16, 32);
    if (half == 0) {
      atomicAdd(&bn_sum[ct * 16 + m], s);
      atomicAdd(&bn_sq[ct * 16 + m], q);
    }
  }
}

// ---------------------------------------------------------------------------
// Edge encoder: ef = relu(edge_attr(16x16,K padded to 32) @ We + be), bf16 out,
// BN stats on the relu output (BN applied later at consumption).
// ---------------------------------------------------------------------------
__global__ __launch_bounds__(256) void edge_enc_kernel(
    const float* __restrict__ ea, const float* __restrict__ We,
    const float* __restrict__ be, __bf16* __restrict__ ef,
    float* bn_sum, float* bn_sq, int nTiles) {
  extern __shared__ char smemraw[];
  __bf16* ldsB = (__bf16*)smemraw;  // 8KB
  fill_B_lds(We, ldsB, 32, 16, 8);
  __syncthreads();

  int wave = threadIdx.x >> 5;
  int lane = threadIdx.x & 31;
  int tile = blockIdx.x * 8 + wave;
  if (tile >= nTiles) return;

  int m = lane & 15;
  int half = lane >> 4;
  int e = tile * 16 + m;
  const float* row = ea + (size_t)e * 16;
  int kh = half * 8;
  v16bf A;
#pragma unroll
  for (int i = 0; i < 8; ++i) {
    A[i]     = (__bf16)row[kh + i];  // real K 0..15
    A[i + 8] = (__bf16)0.0f;        // zero-pad K 16..31
  }

  for (int ct = 0; ct < 8; ++ct) {
    float bv = be[ct * 16 + m];
    v8f c;
#pragma unroll
    for (int r = 0; r < 8; ++r) c[r] = bv;
    c = wmma_bf16(A, loadB(ldsB, 0, ct, 8), c);
    float s = 0.f, q = 0.f;
#pragma unroll
    for (int r = 0; r < 8; ++r) {
      float v = fmaxf(c[r], 0.0f);
      s += v; q += v * v;
      ef[(size_t)(tile * 16 + r + 8 * half) * HID + ct * 16 + m] = (__bf16)v;
    }
    s += __shfl_xor(s, 16, 32);
    q += __shfl_xor(q, 16, 32);
    if (half == 0) {
      atomicAdd(&bn_sum[ct * 16 + m], s);
      atomicAdd(&bn_sq[ct * 16 + m], q);
    }
  }
}

// ---------------------------------------------------------------------------
// Edge MLP stage 1: z1 = relu(concat[h2[src], h2[dst], bn(ef)] @ Wm1 + bm1)
// K = 384 (12 K-tiles). Wm1 (192KB f32) staged via async DMA, swizzled to
// 96KB bf16; total 288KB LDS (<320KB/WGP).
// ---------------------------------------------------------------------------
__global__ __launch_bounds__(256) void edge_mlp1_kernel(
    const __bf16* __restrict__ h2, const __bf16* __restrict__ ef,
    const int* __restrict__ es, const int* __restrict__ et,
    const float* __restrict__ Wm1, const float* __restrict__ bm1,
    const float* __restrict__ escale, const float* __restrict__ eshift,
    __bf16* __restrict__ z1, float* bn_sum, float* bn_sq, int nTiles) {
  extern __shared__ char smemraw[];
  __bf16* ldsB = (__bf16*)smemraw;                 // 12*8*512 bf16 = 96KB
  float*  stage = (float*)(ldsB + (size_t)384 * HID);  // 192KB
  async_stage_to_lds(Wm1, stage, 384 * HID * 4);
  __syncthreads();
  fill_B_lds(stage, ldsB, 384, 384, 8);
  __syncthreads();

  int wave = threadIdx.x >> 5;
  int lane = threadIdx.x & 31;
  int tile = blockIdx.x * 8 + wave;
  if (tile >= nTiles) return;

  int m = lane & 15;
  int half = lane >> 4;
  int e = tile * 16 + m;
  const __bf16* r1 = h2 + (size_t)es[e] * HID;
  const __bf16* r2 = h2 + (size_t)et[e] * HID;
  const __bf16* r3 = ef + (size_t)e * HID;
  __builtin_prefetch(r1, 0, 0);
  __builtin_prefetch(r2, 0, 0);

  v16bf A[12];
#pragma unroll
  for (int kt = 0; kt < 4; ++kt) A[kt] = loadA_bf16(r1, kt);
#pragma unroll
  for (int kt = 0; kt < 4; ++kt) A[4 + kt] = loadA_bf16(r2, kt);
#pragma unroll
  for (int kt = 0; kt < 4; ++kt) A[8 + kt] = loadA_bf16_bn(r3, kt, escale, eshift);

  for (int ct = 0; ct < 8; ++ct) {
    float bv = bm1[ct * 16 + m];
    v8f c;
#pragma unroll
    for (int r = 0; r < 8; ++r) c[r] = bv;
#pragma unroll
    for (int kt = 0; kt < 12; ++kt) c = wmma_bf16(A[kt], loadB(ldsB, kt, ct, 8), c);
    float s = 0.f, q = 0.f;
#pragma unroll
    for (int r = 0; r < 8; ++r) {
      float v = fmaxf(c[r], 0.0f);
      s += v; q += v * v;
      z1[(size_t)(tile * 16 + r + 8 * half) * HID + ct * 16 + m] = (__bf16)v;
    }
    s += __shfl_xor(s, 16, 32);
    q += __shfl_xor(q, 16, 32);
    if (half == 0) {
      atomicAdd(&bn_sum[ct * 16 + m], s);
      atomicAdd(&bn_sq[ct * 16 + m], q);
    }
  }
}

// ---------------------------------------------------------------------------
// Edge MLP stage 2: out = relu(bn(z1) @ Wm2 + bm2) @ Wm3 + bm3
// ---------------------------------------------------------------------------
__global__ __launch_bounds__(256) void edge_mlp2_kernel(
    const __bf16* __restrict__ z1,
    const float* __restrict__ mscale, const float* __restrict__ mshift,
    const float* __restrict__ Wm2, const float* __restrict__ bm2,
    const float* __restrict__ Wm3, const float* __restrict__ bm3,
    float* __restrict__ out, int nTiles) {
  extern __shared__ char smemraw[];
  __bf16* ldsB = (__bf16*)smemraw;  // 4*4*512 bf16 = 16KB
  fill_B_lds(Wm2, ldsB, 128, 128, 4);
  __syncthreads();

  int wave = threadIdx.x >> 5;
  int lane = threadIdx.x & 31;
  int tile = blockIdx.x * 8 + wave;
  if (tile >= nTiles) return;

  int m = lane & 15;
  int half = lane >> 4;
  int e = tile * 16 + m;
  const __bf16* row = z1 + (size_t)e * HID;

  v16bf A[4];
#pragma unroll
  for (int kt = 0; kt < 4; ++kt) A[kt] = loadA_bf16_bn(row, kt, mscale, mshift);

  float p[8];
#pragma unroll
  for (int r = 0; r < 8; ++r) p[r] = 0.0f;

#pragma unroll
  for (int ct = 0; ct < 4; ++ct) {
    float bv = bm2[ct * 16 + m];
    v8f c;
#pragma unroll
    for (int r = 0; r < 8; ++r) c[r] = bv;
#pragma unroll
    for (int kt = 0; kt < 4; ++kt) c = wmma_bf16(A[kt], loadB(ldsB, kt, ct, 4), c);
    float w3 = Wm3[ct * 16 + m];
#pragma unroll
    for (int r = 0; r < 8; ++r) p[r] += fmaxf(c[r], 0.0f) * w3;
  }

  float bm3v = bm3[0];
#pragma unroll
  for (int r = 0; r < 8; ++r) {
    float v = p[r];
    v += __shfl_xor(v, 1, 32);
    v += __shfl_xor(v, 2, 32);
    v += __shfl_xor(v, 4, 32);
    v += __shfl_xor(v, 8, 32);
    p[r] = v;
  }
  if (m == 0) {
#pragma unroll
    for (int r = 0; r < 8; ++r)
      out[(size_t)tile * 16 + r + 8 * half] = p[r] + bm3v;
  }
}

// ---------------------------------------------------------------------------
extern "C" void kernel_launch(void* const* d_in, const int* in_sizes, int n_in,
                              void* d_out, int out_size, void* d_ws, size_t ws_size,
                              hipStream_t stream) {
  const float* x     = (const float*)d_in[0];
  const int*   ei    = (const int*)d_in[1];
  const float* ea    = (const float*)d_in[2];
  const int*   es    = (const int*)d_in[3];
  const int*   et    = (const int*)d_in[4];
  const float* W1l   = (const float*)d_in[5];
  const float* b1    = (const float*)d_in[6];
  const float* W1r   = (const float*)d_in[7];
  const float* W2l   = (const float*)d_in[8];
  const float* b2    = (const float*)d_in[9];
  const float* W2r   = (const float*)d_in[10];
  const float* bn1_g = (const float*)d_in[11];
  const float* bn1_b = (const float*)d_in[12];
  const float* bn2_g = (const float*)d_in[13];
  const float* bn2_b = (const float*)d_in[14];
  const float* We    = (const float*)d_in[15];
  const float* be    = (const float*)d_in[16];
  const float* bne_g = (const float*)d_in[17];
  const float* bne_b = (const float*)d_in[18];
  const float* Wm1   = (const float*)d_in[19];
  const float* bm1   = (const float*)d_in[20];
  const float* bnm_g = (const float*)d_in[21];
  const float* bnm_b = (const float*)d_in[22];
  const float* Wm2   = (const float*)d_in[23];
  const float* bm2   = (const float*)d_in[24];
  const float* Wm3   = (const float*)d_in[25];
  const float* bm3   = (const float*)d_in[26];
  float* out = (float*)d_out;

  const int N = in_sizes[0] / 32;   // 100000
  const int E = in_sizes[3];        // 800000
  const int* srcp = ei;
  const int* dstp = ei + E;

  // ---- workspace carving ----
  char* w = (char*)d_ws;
  auto carve = [&](size_t bytes) {
    char* p = w;
    w += (bytes + 255) & ~(size_t)255;
    return p;
  };
  float*  stats = (float*)carve(4 * 512 * sizeof(float));
  float*  cnt   = (float*)carve((size_t)N * sizeof(float));
  float*  h1    = (float*)carve((size_t)N * HID * sizeof(float));
  float*  poolA = (float*)carve((size_t)N * HID * sizeof(float)); // h1pre -> agg2
  float*  poolB = (float*)carve((size_t)N * HID * sizeof(float)); // agg1 -> h2pre
  __bf16* h2bf  = (__bf16*)carve((size_t)N * HID * sizeof(__bf16));
  __bf16* ef    = (__bf16*)carve((size_t)E * HID * sizeof(__bf16));
  __bf16* z1    = (__bf16*)carve((size_t)E * HID * sizeof(__bf16));

  auto S  = [&](int i) { return stats + (size_t)i * 512; };        // sum
  auto Q  = [&](int i) { return stats + (size_t)i * 512 + 128; };  // sumsq
  auto SC = [&](int i) { return stats + (size_t)i * 512 + 256; };  // scale
  auto SH = [&](int i) { return stats + (size_t)i * 512 + 384; };  // shift
  // 0=bn1  1=bn2  2=bne  3=bnm

  const int nTilesN = N / 16;   // 6250
  const int nTilesE = E / 16;   // 50000
  const int blkN = (nTilesN + 7) / 8;
  const int blkE = (nTilesE + 7) / 8;

  const size_t lds_sage32  = (size_t)2 * 32 * HID * sizeof(__bf16);            // 16KB
  const size_t lds_sage128 = (size_t)2 * 128 * HID * (sizeof(__bf16) + 4);     // 64KB + 128KB stage
  const size_t lds_enc     = (size_t)32 * HID * sizeof(__bf16);                //  8KB
  const size_t lds_mlp1    = (size_t)384 * HID * (sizeof(__bf16) + 4);         // 96KB + 192KB stage
  const size_t lds_mlp2    = (size_t)128 * 64 * sizeof(__bf16);                // 16KB

  (void)hipFuncSetAttribute((const void*)sage_gemm_kernel<128>,
      hipFuncAttributeMaxDynamicSharedMemorySize, (int)lds_sage128);
  (void)hipFuncSetAttribute((const void*)edge_mlp1_kernel,
      hipFuncAttributeMaxDynamicSharedMemorySize, (int)lds_mlp1);

  // ---- stage 0: init ----
  zero_kernel<<<64, 256, 0, stream>>>(stats, 4 * 512);
  zero_kernel<<<256, 256, 0, stream>>>(cnt, N);
  zero_kernel<<<1024, 256, 0, stream>>>(poolB, (long)N * 32);       // agg1
  count_kernel<<<(E + 255) / 256, 256, 0, stream>>>(dstp, cnt, E);

  // ---- layer 1 ----
  scatter_add_kernel<32><<<2048, 256, 0, stream>>>(x, srcp, dstp, poolB, E);
  sage_gemm_kernel<32><<<blkN, 256, lds_sage32, stream>>>(
      poolB, cnt, x, W1l, b1, W1r, poolA, S(0), Q(0), nTilesN);
  bn_finalize_kernel<<<1, 128, 0, stream>>>(S(0), Q(0), bn1_g, bn1_b, SC(0), SH(0), (float)N);
  bn_apply_relu_kernel<float><<<2048, 256, 0, stream>>>(poolA, SC(0), SH(0), h1, (long)N * HID);

  // ---- layer 2 ----
  zero_kernel<<<2048, 256, 0, stream>>>(poolA, (long)N * HID);      // agg2
  scatter_add_kernel<128><<<4096, 256, 0, stream>>>(h1, srcp, dstp, poolA, E);
  sage_gemm_kernel<128><<<blkN, 256, lds_sage128, stream>>>(
      poolA, cnt, h1, W2l, b2, W2r, poolB, S(1), Q(1), nTilesN);
  bn_finalize_kernel<<<1, 128, 0, stream>>>(S(1), Q(1), bn2_g, bn2_b, SC(1), SH(1), (float)N);
  bn_apply_relu_kernel<__bf16><<<2048, 256, 0, stream>>>(poolB, SC(1), SH(1), h2bf, (long)N * HID);

  // ---- edge encoder ----
  edge_enc_kernel<<<blkE, 256, lds_enc, stream>>>(ea, We, be, ef, S(2), Q(2), nTilesE);
  bn_finalize_kernel<<<1, 128, 0, stream>>>(S(2), Q(2), bne_g, bne_b, SC(2), SH(2), (float)E);

  // ---- edge MLP ----
  edge_mlp1_kernel<<<blkE, 256, lds_mlp1, stream>>>(
      h2bf, ef, es, et, Wm1, bm1, SC(2), SH(2), z1, S(3), Q(3), nTilesE);
  bn_finalize_kernel<<<1, 128, 0, stream>>>(S(3), Q(3), bnm_g, bnm_b, SC(3), SH(3), (float)E);
  edge_mlp2_kernel<<<blkE, 256, lds_mlp2, stream>>>(
      z1, SC(3), SH(3), Wm2, bm2, Wm3, bm3, out, nTilesE);
}
